// step_StateBayesianKalmanNet_11862699672213
// MI455X (gfx1250) — matrix operand order repaired and checked by reading, not compiled, not run on hardware
//
#include <hip/hip_runtime.h>
#include <hip/hip_bf16.h>
#include <math.h>

#define S_DIM 16
#define O_DIM 8
#define J_DIM 32
#define B_DIM 4096
#define HID 240
#define HIDP 256           // K padded to multiple of 32
#define OUT_HID 512
#define M_TOT (J_DIM * B_DIM)   // 131072
#define N_TILES_M (M_TOT / 16)  // 8192

typedef __attribute__((ext_vector_type(16))) __bf16 v16bf;
typedef __attribute__((ext_vector_type(8)))  __bf16 v8bf;
typedef __attribute__((ext_vector_type(8)))  float  v8f;

// round-to-nearest-even f32 -> bf16
__device__ __forceinline__ __bf16 f2bf(float f) {
    unsigned u = __builtin_bit_cast(unsigned, f);
    u += 0x7fffu + ((u >> 16) & 1u);
    unsigned short h = (unsigned short)(u >> 16);
    return __builtin_bit_cast(__bf16, h);
}

// A fragment: rowp = &A[row][k0]; kb = (lane>>4)*8
// elems 0..7  = A[row][k0+kb .. k0+kb+7]
// elems 8..15 = A[row][k0+kb+16 .. k0+kb+23]
__device__ __forceinline__ v16bf ldfragA(const __bf16* rowp, int kb) {
    const __bf16* p = rowp + kb;
    v8bf lo = *(const v8bf*)(p);
    v8bf hi = *(const v8bf*)(p + 16);
    return __builtin_shufflevector(lo, hi, 0,1,2,3,4,5,6,7,8,9,10,11,12,13,14,15);
}

// B fragment: p = &W[n][k0 + (lane>>4)*8]; elems 0..7 = K p..p+7, elems 8..15 = K p+16..p+23
__device__ __forceinline__ v16bf ldfragB(const __bf16* p) {
    v8bf lo = *(const v8bf*)(p);
    v8bf hi = *(const v8bf*)(p + 16);
    return __builtin_shufflevector(lo, hi, 0,1,2,3,4,5,6,7,8,9,10,11,12,13,14,15);
}

#define WMMA_BF16(a, b, c) \
    __builtin_amdgcn_wmma_f32_16x16x32_bf16(false, (a), false, (b), (short)0, (c), false, false)

// ---------------- prep kernels ----------------

__global__ void k_xpred(const float* __restrict__ xprev, const float* __restrict__ F,
                        float* __restrict__ xpred) {
    int idx = blockIdx.x * blockDim.x + threadIdx.x;
    if (idx >= B_DIM * S_DIM) return;
    int b = idx >> 4, s = idx & 15;
    float acc = 0.f;
    #pragma unroll
    for (int k = 0; k < S_DIM; ++k) acc += xprev[b * S_DIM + k] * F[s * S_DIM + k];
    xpred[idx] = acc;
}

__global__ void k_inno(const float* __restrict__ y, const float* __restrict__ xpred,
                       const float* __restrict__ H, float* __restrict__ inno) {
    int idx = blockIdx.x * blockDim.x + threadIdx.x;
    if (idx >= B_DIM * O_DIM) return;
    int b = idx >> 3, o = idx & 7;
    float acc = 0.f;
    #pragma unroll
    for (int s = 0; s < S_DIM; ++s) acc += xpred[b * S_DIM + s] * H[o * S_DIM + s];
    inno[idx] = y[idx] - acc;
}

__global__ void k_xin(const float* __restrict__ xprev, const float* __restrict__ xpred,
                      const float* __restrict__ inno, const float* __restrict__ Wfc,
                      const float* __restrict__ bfc, float* __restrict__ xin) {
    int idx = blockIdx.x * blockDim.x + threadIdx.x;
    if (idx >= B_DIM * HID) return;
    int b = idx / HID, h = idx % HID;
    const float* w = Wfc + h * 48;
    float acc = bfc[h];
    #pragma unroll
    for (int k = 0; k < S_DIM; ++k)
        acc += (xprev[b * S_DIM + k] - xpred[b * S_DIM + k]) * w[k];
    #pragma unroll
    for (int o = 0; o < O_DIM; ++o)
        acc += inno[b * O_DIM + o] * (w[16 + o] + w[40 + o]);  // diff_state cols are zero
    xin[idx] = acc > 0.f ? acc : 0.f;
}

__global__ void k_gh(const float* __restrict__ hinit, const float* __restrict__ Whh,
                     const float* __restrict__ bhh, float* __restrict__ gh) {
    int n = blockIdx.x * blockDim.x + threadIdx.x;
    if (n >= 3 * HID) return;
    float acc = bhh[n];
    const float* w = Whh + n * HID;
    for (int k = 0; k < HID; ++k) acc += hinit[k] * w[k];
    gh[n] = acc;
}

__global__ void k_wcvt(const float* __restrict__ Wih, const float* __restrict__ Wo1,
                       const float* __restrict__ Wo2,
                       unsigned short* __restrict__ dWih_u, unsigned short* __restrict__ dWo1_u,
                       unsigned short* __restrict__ dWo2_u) {
    __bf16* dWih = reinterpret_cast<__bf16*>(dWih_u);
    __bf16* dWo1 = reinterpret_cast<__bf16*>(dWo1_u);
    __bf16* dWo2 = reinterpret_cast<__bf16*>(dWo2_u);
    const int N0 = 3 * HID * HIDP;      // 184320
    const int N1 = OUT_HID * HIDP;      // 131072
    const int N2 = S_DIM * O_DIM * OUT_HID; // 65536
    int idx = blockIdx.x * blockDim.x + threadIdx.x;
    if (idx < N0) {
        int n = idx >> 8, k = idx & 255;
        dWih[idx] = (k < HID) ? f2bf(Wih[n * HID + k]) : f2bf(0.f);
    } else if (idx < N0 + N1) {
        int i = idx - N0;
        int n = i >> 8, k = i & 255;
        dWo1[i] = (k < HID) ? f2bf(Wo1[n * HID + k]) : f2bf(0.f);
    } else if (idx < N0 + N1 + N2) {
        int i = idx - N0 - N1;
        dWo2[i] = f2bf(Wo2[i]);
    }
}

// ---------------- fused main kernel: one wave per 16-row M tile ----------------

__global__ __launch_bounds__(32)
void k_fused(const float* __restrict__ xin, const float* __restrict__ u1,
             const float* __restrict__ u2, const float* __restrict__ gh,
             const float* __restrict__ bih, const float* __restrict__ hinit,
             const float* __restrict__ bo1, const float* __restrict__ bo2,
             const unsigned short* __restrict__ Wih_u, const unsigned short* __restrict__ Wo1_u,
             const unsigned short* __restrict__ Wo2_u, const float* __restrict__ inno,
             const float* __restrict__ xpred, float* __restrict__ ens) {
    const __bf16* Wih = reinterpret_cast<const __bf16*>(Wih_u);
    const __bf16* Wo1 = reinterpret_cast<const __bf16*>(Wo1_u);
    const __bf16* Wo2 = reinterpret_cast<const __bf16*>(Wo2_u);

    __shared__ __align__(16) __bf16 A1[16][HIDP];     // masked x_in (bf16)
    __shared__ __align__(16) __bf16 A2[16][HIDP];     // h_new (bf16)
    __shared__ __align__(16) __bf16 A3[16][OUT_HID];  // masked o1 (bf16)
    __shared__ float Kt[16][128];                     // K gain tile
    __shared__ float innoT[16][O_DIM];
    __shared__ float xpT[16][S_DIM];

    const int l = threadIdx.x;
    const int lr = l & 15, lh = l >> 4;
    const int kbA = lh * 8;     // A fragment K base
    const int kbB = lh * 8;     // B fragment K base
    const int tile = blockIdx.x;
    const int j = tile >> 8;                 // B/16 = 256 tiles per j
    const int b0 = (tile & 255) << 4;
    const int m0 = tile << 4;                // = j*B + b0

    // ---- stage inputs: A1 = bf16(x_in * mask1), zero pads, inno/xpred tiles ----
    for (int row = 0; row < 16; ++row) {
        for (int h = l; h < HID; h += 32) {
            float x = xin[(b0 + row) * HID + h];
            float u = u1[(size_t)(m0 + row) * HID + h];
            A1[row][h] = f2bf(u > 0.5f ? 2.0f * x : 0.0f);
        }
    }
    for (int i = l; i < 16 * 16; i += 32) {
        A1[i >> 4][HID + (i & 15)] = f2bf(0.f);
        A2[i >> 4][HID + (i & 15)] = f2bf(0.f);
    }
    for (int i = l; i < 16 * O_DIM; i += 32) innoT[i >> 3][i & 7] = inno[b0 * O_DIM + i];
    for (int i = l; i < 16 * S_DIM; i += 32) xpT[i >> 4][i & 15] = xpred[b0 * S_DIM + i];
    __syncthreads();

    // ---- GEMM1 (xd @ W_ih.T) fused with GRU cell -> A2 = bf16(h_new) ----
    for (int t2 = 0; t2 < HID / 16; ++t2) {          // 15 gate-column tiles
        v8f aR = {}, aZ = {}, aN = {};
        const int nR = t2 * 16 + lr;
        const __bf16* wR = Wih + (size_t)nR * HIDP;
        const __bf16* wZ = Wih + (size_t)(HID + nR) * HIDP;
        const __bf16* wN = Wih + (size_t)(2 * HID + nR) * HIDP;
        #pragma unroll
        for (int ks = 0; ks < HIDP / 32; ++ks) {
            const int k0 = ks * 32;
            v16bf a  = ldfragA(&A1[lr][k0], kbA);
            v16bf bR = ldfragB(wR + k0 + kbB);
            v16bf bZ = ldfragB(wZ + k0 + kbB);
            v16bf bN = ldfragB(wN + k0 + kbB);
            aR = WMMA_BF16(a, bR, aR);
            aZ = WMMA_BF16(a, bZ, aZ);
            aN = WMMA_BF16(a, bN, aN);
        }
        const int hcol = t2 * 16 + lr;
        const float hr = gh[hcol], hz = gh[HID + hcol], hn = gh[2 * HID + hcol];
        const float br = bih[hcol], bz = bih[HID + hcol], bn = bih[2 * HID + hcol];
        const float hi = hinit[hcol];
        #pragma unroll
        for (int v = 0; v < 8; ++v) {
            float r = 1.f / (1.f + __expf(-(aR[v] + br + hr)));
            float z = 1.f / (1.f + __expf(-(aZ[v] + bz + hz)));
            float n = tanhf(aN[v] + bn + r * hn);
            float hv = (1.f - z) * n + z * hi;
            A2[v + 8 * lh][hcol] = f2bf(hv);
        }
    }
    __syncthreads();

    // ---- GEMM2 (h_new @ W_out1.T) + bias + ReLU + mask2 -> A3 ----
    for (int t = 0; t < OUT_HID / 16; ++t) {         // 32 column tiles
        v8f acc = {};
        const int n = t * 16 + lr;
        const __bf16* w = Wo1 + (size_t)n * HIDP;
        #pragma unroll
        for (int ks = 0; ks < HIDP / 32; ++ks) {
            const int k0 = ks * 32;
            v16bf a = ldfragA(&A2[lr][k0], kbA);
            v16bf b = ldfragB(w + k0 + kbB);
            acc = WMMA_BF16(a, b, acc);
        }
        const float bias = bo1[n];
        #pragma unroll
        for (int v = 0; v < 8; ++v) {
            const int M = v + 8 * lh;
            float o = acc[v] + bias;
            o = o > 0.f ? o : 0.f;
            float u = u2[(size_t)(m0 + M) * OUT_HID + n];
            A3[M][n] = f2bf(u > 0.65f ? o * (1.0f / 0.35f) : 0.0f);
        }
    }
    __syncthreads();

    // ---- GEMM3 (o1 @ W_out2.T) + bias -> Kt ----
    for (int t = 0; t < 8; ++t) {                    // 128 output columns
        v8f acc = {};
        const int n = t * 16 + lr;
        const __bf16* w = Wo2 + (size_t)n * OUT_HID;
        #pragma unroll
        for (int ks = 0; ks < OUT_HID / 32; ++ks) {
            const int k0 = ks * 32;
            v16bf a = ldfragA(&A3[lr][k0], kbA);
            v16bf b = ldfragB(w + k0 + kbB);
            acc = WMMA_BF16(a, b, acc);
        }
        const float bias = bo2[n];
        #pragma unroll
        for (int v = 0; v < 8; ++v) Kt[v + 8 * lh][n] = acc[v] + bias;
    }
    __syncthreads();

    // ---- Kalman correction + ensemble write: ens[b][j][s] ----
    for (int idx = l; idx < 16 * S_DIM; idx += 32) {
        const int row = idx >> 4, s = idx & 15;
        float sum = 0.f;
        #pragma unroll
        for (int o = 0; o < O_DIM; ++o) sum += Kt[row][s * O_DIM + o] * innoT[row][o];
        ens[(size_t)(b0 + row) * (J_DIM * S_DIM) + j * S_DIM + s] = xpT[row][s] + sum;
    }
}

// ---------------- ensemble reduction: mean + unbiased var over J ----------------

__global__ void k_reduce(const float* __restrict__ ens, float* __restrict__ out) {
    int idx = blockIdx.x * blockDim.x + threadIdx.x;
    if (idx >= B_DIM * S_DIM) return;
    int b = idx >> 4, s = idx & 15;
    const float* p = ens + (size_t)b * (J_DIM * S_DIM) + s;
    float vals[J_DIM];
    float sum = 0.f;
    #pragma unroll
    for (int jj = 0; jj < J_DIM; ++jj) { vals[jj] = p[jj * S_DIM]; sum += vals[jj]; }
    float mean = sum * (1.0f / J_DIM);
    float var = 0.f;
    #pragma unroll
    for (int jj = 0; jj < J_DIM; ++jj) { float d = vals[jj] - mean; var += d * d; }
    out[idx] = mean;
    out[B_DIM * S_DIM + idx] = var * (1.0f / (J_DIM - 1));
    if (idx == 0) {
        out[2 * B_DIM * S_DIM] =
            -(0.5f * __logf(0.5f) + 0.5f * __logf(0.5f))
            - (0.65f * __logf(0.65f) + 0.35f * __logf(0.35f));
    }
}

// ---------------- launch ----------------

extern "C" void kernel_launch(void* const* d_in, const int* in_sizes, int n_in,
                              void* d_out, int out_size, void* d_ws, size_t ws_size,
                              hipStream_t stream) {
    const float* y_t   = (const float*)d_in[0];
    const float* xprev = (const float*)d_in[1];
    const float* F     = (const float*)d_in[2];
    const float* H     = (const float*)d_in[3];
    const float* Wfc   = (const float*)d_in[4];
    const float* bfc   = (const float*)d_in[5];
    const float* Wih   = (const float*)d_in[6];
    const float* Whh   = (const float*)d_in[7];
    const float* bih   = (const float*)d_in[8];
    const float* bhh   = (const float*)d_in[9];
    const float* Wo1   = (const float*)d_in[10];
    const float* bo1   = (const float*)d_in[11];
    const float* Wo2   = (const float*)d_in[12];
    const float* bo2   = (const float*)d_in[13];
    const float* hinit = (const float*)d_in[14];
    const float* u1    = (const float*)d_in[15];
    const float* u2    = (const float*)d_in[16];
    float* out = (float*)d_out;

    char* ws = (char*)d_ws;
    float* xpred = (float*)(ws + 0);              // B*16*4       = 262144
    float* inno  = (float*)(ws + 262144);         // B*8*4        = 131072
    float* xin   = (float*)(ws + 393216);         // B*240*4      = 3932160
    float* gh    = (float*)(ws + 4325376);        // 720*4        (padded)
    unsigned short* dWih = (unsigned short*)(ws + 4328448);  // 720*256*2 = 368640
    unsigned short* dWo1 = (unsigned short*)(ws + 4697088);  // 512*256*2 = 262144
    unsigned short* dWo2 = (unsigned short*)(ws + 4959232);  // 128*512*2 = 131072

    float* ens = out + 2 * B_DIM * S_DIM + 1;     // ensemble region of d_out

    k_xpred<<<(B_DIM * S_DIM + 255) / 256, 256, 0, stream>>>(xprev, F, xpred);
    k_inno <<<(B_DIM * O_DIM + 255) / 256, 256, 0, stream>>>(y_t, xpred, H, inno);
    k_xin  <<<(B_DIM * HID + 255) / 256, 256, 0, stream>>>(xprev, xpred, inno, Wfc, bfc, xin);
    k_gh   <<<3, 256, 0, stream>>>(hinit, Whh, bhh, gh);
    {
        const int total = 3 * HID * HIDP + OUT_HID * HIDP + S_DIM * O_DIM * OUT_HID;
        k_wcvt<<<(total + 255) / 256, 256, 0, stream>>>(Wih, Wo1, Wo2, dWih, dWo1, dWo2);
    }
    k_fused<<<N_TILES_M, 32, 0, stream>>>(xin, u1, u2, gh, bih, hinit, bo1, bo2,
                                          dWih, dWo1, dWo2, inno, xpred, ens);
    k_reduce<<<(B_DIM * S_DIM + 255) / 256, 256, 0, stream>>>(ens, out);
}
